// MemoryModule_2568390443082
// MI455X (gfx1250) — matrix operand order repaired and checked by reading, not compiled, not run on hardware
//
#include <hip/hip_runtime.h>
#include <cstdint>

#define NUM_B   65536
#define DIM     512
#define NSLOT   2048
#define BM      64          // rows per block: 4 M-tiles share every B fragment
#define MT      4           // M-tiles per block
#define THREADS 512         // 16 waves

typedef __bf16 bf16_t;
typedef __bf16 v4bf  __attribute__((ext_vector_type(4)));
typedef __bf16 v8bf  __attribute__((ext_vector_type(8)));
typedef __bf16 v16bf __attribute__((ext_vector_type(16)));
typedef float  v8f   __attribute__((ext_vector_type(8)));

static __device__ __forceinline__ v16bf cat16(v8bf lo, v8bf hi) {
  return __builtin_shufflevector(lo, hi, 0,1,2,3,4,5,6,7,8,9,10,11,12,13,14,15);
}

#define WMMA_BF16(A, B, C) \
  __builtin_amdgcn_wmma_f32_16x16x32_bf16(false, (A), false, (B), (short)0, (C), false, false)

// ---------------------------------------------------------------------------
// Prep: L2-normalize memory rows -> bf16 row-major (GEMM1 B operand), and
// store unnormalized memory transposed -> bf16 (GEMM2 B operand). 4 MB total,
// L2-resident for the whole main kernel.
// ---------------------------------------------------------------------------
__global__ void __launch_bounds__(64) prep_kernel(const float* __restrict__ mem,
                                                  bf16_t* __restrict__ mem_nbf,
                                                  bf16_t* __restrict__ memT) {
  __shared__ float red[64];
  const int n = blockIdx.x;
  const int t = threadIdx.x;                     // 0..63
  const float4* row = (const float4*)(mem + (size_t)n * DIM);
  float4 a0 = row[t];
  float4 a1 = row[t + 64];
  float ss = a0.x*a0.x + a0.y*a0.y + a0.z*a0.z + a0.w*a0.w
           + a1.x*a1.x + a1.y*a1.y + a1.z*a1.z + a1.w*a1.w;
  red[t] = ss;
  __syncthreads();
  for (int s = 32; s > 0; s >>= 1) {
    if (t < s) red[t] += red[t + s];
    __syncthreads();
  }
  const float scale = 1.0f / fmaxf(sqrtf(red[0]), 1e-12f);
  const int d0 = t * 4, d1 = (t + 64) * 4;
  bf16_t* on = mem_nbf + (size_t)n * DIM;
  v4bf b0 = { (bf16_t)(a0.x*scale), (bf16_t)(a0.y*scale),
              (bf16_t)(a0.z*scale), (bf16_t)(a0.w*scale) };
  v4bf b1 = { (bf16_t)(a1.x*scale), (bf16_t)(a1.y*scale),
              (bf16_t)(a1.z*scale), (bf16_t)(a1.w*scale) };
  *(v4bf*)(on + d0) = b0;
  *(v4bf*)(on + d1) = b1;
  memT[(size_t)(d0+0)*NSLOT + n] = (bf16_t)a0.x;
  memT[(size_t)(d0+1)*NSLOT + n] = (bf16_t)a0.y;
  memT[(size_t)(d0+2)*NSLOT + n] = (bf16_t)a0.z;
  memT[(size_t)(d0+3)*NSLOT + n] = (bf16_t)a0.w;
  memT[(size_t)(d1+0)*NSLOT + n] = (bf16_t)a1.x;
  memT[(size_t)(d1+1)*NSLOT + n] = (bf16_t)a1.y;
  memT[(size_t)(d1+2)*NSLOT + n] = (bf16_t)a1.z;
  memT[(size_t)(d1+3)*NSLOT + n] = (bf16_t)a1.w;
}

// ---------------------------------------------------------------------------
// Fused per 64 query rows: normalize z -> GEMM1 (WMMA bf16, B shared by 4
// M-tiles) -> softmax -> attn written once -> GEMM2 (WMMA bf16) -> z_hat.
// LDS: exp/logits bf16 [64][2048] = 256 KB + z bf16 [64][512] = 64 KB
//      = 320 KB (full CDNA5 WGP LDS). Scratch overlays dead regions.
// L2 B-operand traffic: 1024 blocks x 4 MB = 4 GB (4x less than BM=16).
// ---------------------------------------------------------------------------
__global__ void __launch_bounds__(THREADS) attn_kernel(
    const float* __restrict__ z,
    const bf16_t* __restrict__ mem_nbf,
    const bf16_t* __restrict__ memT,
    float* __restrict__ zhat,
    float* __restrict__ attn)
{
  extern __shared__ char smem_raw[];
  bf16_t* lgb = (bf16_t*)smem_raw;            // [BM][NSLOT] bf16 logits/exp, 256 KB
  bf16_t* zbf = lgb + (size_t)BM * NSLOT;     // [BM][DIM]   bf16 z_norm,     64 KB
  // Phase-1 scratch overlays logits (not yet written):
  float* scrA     = (float*)lgb;              // [BM][8] partial sumsq
  float* rowscale = scrA + 512;               // [BM]
  // Softmax scratch overlays zbf (dead after GEMM1):
  float* scrB   = (float*)zbf;                // [BM][8] partials
  float* rowmax = scrB + 512;                 // [BM]
  float* rowinv = scrB + 576;                 // [BM]

  const int tid  = threadIdx.x;
  const int row0 = blockIdx.x * BM;
  const int wave = tid >> 5;                  // 0..15
  const int lane = tid & 31;
  const int lm   = lane & 15;                 // M row (A) / N col (B) within tile
  const int kh   = lane >> 4;                 // K-half selector (WMMA layout)

  // ---------------- Phase 1: load + L2-normalize 64 z rows -> bf16 LDS ----
  {
    const int r  = tid >> 3;                  // row 0..63, 8 threads per row
    const int c8 = tid & 7;
    const float4* zr = (const float4*)(z + ((size_t)row0 + r) * DIM);
    float4 v[16];
    float ss = 0.f;
#pragma unroll
    for (int j = 0; j < 16; ++j) {
      v[j] = zr[c8 + j * 8];
      ss += v[j].x*v[j].x + v[j].y*v[j].y + v[j].z*v[j].z + v[j].w*v[j].w;
    }
    scrA[tid] = ss;
    __syncthreads();
    if (tid < BM) {
      float s = 0.f;
#pragma unroll
      for (int j = 0; j < 8; ++j) s += scrA[tid * 8 + j];
      rowscale[tid] = 1.0f / fmaxf(sqrtf(s), 1e-12f);
    }
    __syncthreads();
    const float zs = rowscale[r];
#pragma unroll
    for (int j = 0; j < 16; ++j) {
      float4 x = v[j];
      v4bf b = { (bf16_t)(x.x*zs), (bf16_t)(x.y*zs),
                 (bf16_t)(x.z*zs), (bf16_t)(x.w*zs) };
      *(v4bf*)(zbf + r * DIM + (c8 + j * 8) * 4) = b;
    }
  }
  __syncthreads();

  // ---------------- Phase 2: GEMM1 logits = z_norm @ mem_norm^T -----------
  // Each wave: 8 N-tiles (in pairs), all 4 M-tiles; every B fragment feeds
  // 4 WMMAs -> mem_nbf is read exactly once per block.
  {
    for (int ntg = 0; ntg < 4; ++ntg) {
      const int nt0 = wave * 8 + ntg * 2;
      const bf16_t* bp0 = mem_nbf + (size_t)(nt0 * 16 + lm) * DIM;
      const bf16_t* bp1 = bp0 + (size_t)16 * DIM;
      v8f c0[MT], c1[MT];
#pragma unroll
      for (int mt = 0; mt < MT; ++mt) {
        c0[mt] = (v8f){0.f,0.f,0.f,0.f,0.f,0.f,0.f,0.f};
        c1[mt] = (v8f){0.f,0.f,0.f,0.f,0.f,0.f,0.f,0.f};
      }
#pragma unroll 4
      for (int s = 0; s < 16; ++s) {
        v16bf A[MT];
#pragma unroll
        for (int mt = 0; mt < MT; ++mt) {
          const bf16_t* pa = zbf + (mt * 16 + lm) * DIM + s * 32 + kh * 16;
          A[mt] = cat16(*(const v8bf*)pa, *(const v8bf*)(pa + 8));
        }
        const bf16_t* pb0 = bp0 + s * 32 + kh * 16;
        const bf16_t* pb1 = bp1 + s * 32 + kh * 16;
        v16bf B0 = cat16(*(const v8bf*)pb0, *(const v8bf*)(pb0 + 8));
        v16bf B1 = cat16(*(const v8bf*)pb1, *(const v8bf*)(pb1 + 8));
#pragma unroll
        for (int mt = 0; mt < MT; ++mt) {
          c0[mt] = WMMA_BF16(A[mt], B0, c0[mt]);
          c1[mt] = WMMA_BF16(A[mt], B1, c1[mt]);
        }
      }
#pragma unroll
      for (int mt = 0; mt < MT; ++mt)
#pragma unroll
        for (int rr = 0; rr < 8; ++rr) {
          const int mrow = mt * 16 + rr + kh * 8;
          lgb[mrow * NSLOT + nt0 * 16 + lm]       = (bf16_t)c0[mt][rr];
          lgb[mrow * NSLOT + (nt0 + 1) * 16 + lm] = (bf16_t)c1[mt][rr];
        }
    }
  }
  __syncthreads();

  // ---------------- Phase 3: softmax (max, exp, sum) on bf16 logits -------
  {
    const int r  = tid >> 3;
    const int c8 = tid & 7;
    v4bf* lr = (v4bf*)(lgb + r * NSLOT);      // 512 v4bf per row
    float mx = -1e30f;
#pragma unroll 4
    for (int j = 0; j < 64; ++j) {
      v4bf x = lr[c8 + j * 8];
      mx = fmaxf(mx, fmaxf(fmaxf((float)x[0], (float)x[1]),
                           fmaxf((float)x[2], (float)x[3])));
    }
    scrB[tid] = mx;
    __syncthreads();
    if (tid < BM) {
      float m = -1e30f;
#pragma unroll
      for (int j = 0; j < 8; ++j) m = fmaxf(m, scrB[tid * 8 + j]);
      rowmax[tid] = m;
    }
    __syncthreads();
    const float rm = rowmax[r];
    float sum = 0.f;
#pragma unroll 4
    for (int j = 0; j < 64; ++j) {
      v4bf x = lr[c8 + j * 8];
      v4bf e;
      e[0] = (bf16_t)__expf((float)x[0] - rm);
      e[1] = (bf16_t)__expf((float)x[1] - rm);
      e[2] = (bf16_t)__expf((float)x[2] - rm);
      e[3] = (bf16_t)__expf((float)x[3] - rm);
      lr[c8 + j * 8] = e;
      sum += (float)e[0] + (float)e[1] + (float)e[2] + (float)e[3];
    }
    scrB[tid] = sum;
    __syncthreads();
    if (tid < BM) {
      float s = 0.f;
#pragma unroll
      for (int j = 0; j < 8; ++j) s += scrB[tid * 8 + j];
      rowinv[tid] = 1.0f / s;
    }
    __syncthreads();
  }

  // ---------------- Phase 4: write attn (coalesced float4, written once) --
  {
    const v4bf* l4 = (const v4bf*)lgb;
    float4* ag4 = (float4*)(attn + (size_t)row0 * NSLOT);
#pragma unroll 4
    for (int i = 0; i < 64; ++i) {
      const int q  = tid + i * 512;           // 32768 v4bf per block
      const int rr = q >> 9;                  // 512 v4bf per row
      const float s = rowinv[rr];
      v4bf x = l4[q];
      ag4[q] = make_float4((float)x[0]*s, (float)x[1]*s,
                           (float)x[2]*s, (float)x[3]*s);
    }
  }

  // ---------------- Phase 5: GEMM2 z_hat = attn @ memory ------------------
  // Each wave: 2 D-tiles, all 4 M-tiles; B fragment feeds 4 WMMAs.
  // A = stored bf16 exp values; 1/rowsum folded into the f32 epilogue.
  {
    const int dt0 = wave * 2;
    const bf16_t* bt0 = memT + (size_t)(dt0 * 16 + lm) * NSLOT;
    const bf16_t* bt1 = bt0 + (size_t)16 * NSLOT;
    v8f a0[MT], a1[MT];
#pragma unroll
    for (int mt = 0; mt < MT; ++mt) {
      a0[mt] = (v8f){0.f,0.f,0.f,0.f,0.f,0.f,0.f,0.f};
      a1[mt] = (v8f){0.f,0.f,0.f,0.f,0.f,0.f,0.f,0.f};
    }
#pragma unroll 2
    for (int s = 0; s < 64; ++s) {            // K = 2048
      v16bf A[MT];
#pragma unroll
      for (int mt = 0; mt < MT; ++mt) {
        const bf16_t* pa = lgb + (mt * 16 + lm) * NSLOT + s * 32 + kh * 16;
        A[mt] = cat16(*(const v8bf*)pa, *(const v8bf*)(pa + 8));
      }
      const bf16_t* pb0 = bt0 + s * 32 + kh * 16;
      const bf16_t* pb1 = bt1 + s * 32 + kh * 16;
      v16bf B0 = cat16(*(const v8bf*)pb0, *(const v8bf*)(pb0 + 8));
      v16bf B1 = cat16(*(const v8bf*)pb1, *(const v8bf*)(pb1 + 8));
#pragma unroll
      for (int mt = 0; mt < MT; ++mt) {
        a0[mt] = WMMA_BF16(A[mt], B0, a0[mt]);
        a1[mt] = WMMA_BF16(A[mt], B1, a1[mt]);
      }
    }
#pragma unroll
    for (int mt = 0; mt < MT; ++mt)
#pragma unroll
      for (int rr = 0; rr < 8; ++rr) {
        const int mrow = mt * 16 + rr + kh * 8;
        const float s = rowinv[mrow];
        float* zr = zhat + ((size_t)row0 + mrow) * DIM;
        zr[dt0 * 16 + lm]       = a0[mt][rr] * s;
        zr[(dt0 + 1) * 16 + lm] = a1[mt][rr] * s;
      }
  }
}

extern "C" void kernel_launch(void* const* d_in, const int* in_sizes, int n_in,
                              void* d_out, int out_size, void* d_ws, size_t ws_size,
                              hipStream_t stream) {
  const float* z   = (const float*)d_in[0];
  const float* mem = (const float*)d_in[1];

  bf16_t* mem_nbf = (bf16_t*)d_ws;                       // [NSLOT][DIM] bf16, normalized
  bf16_t* memT    = mem_nbf + (size_t)NSLOT * DIM;       // [DIM][NSLOT] bf16, raw, transposed

  float* zhat = (float*)d_out;                           // [NUM_B][DIM]
  float* attn = zhat + (size_t)NUM_B * DIM;              // [NUM_B][NSLOT]

  prep_kernel<<<NSLOT, 64, 0, stream>>>(mem, mem_nbf, memT);

  const size_t smem = (size_t)BM * NSLOT * sizeof(bf16_t)   // 256 KB logits/exp
                    + (size_t)BM * DIM * sizeof(bf16_t);    //  64 KB z bf16 / scratch
  attn_kernel<<<NUM_B / BM, THREADS, smem, stream>>>(z, mem_nbf, memT, zhat, attn);
}
